// MultiHeadAttention_50697793962112
// MI455X (gfx1250) — matrix operand order repaired and checked by reading, not compiled, not run on hardware
//
#include <hip/hip_runtime.h>
#include <hip/hip_bf16.h>

// ---- problem constants (match reference) ----
#define Bb 8
#define Nn 1024
#define Cc 1024
#define Hh 16
#define Dd 64
#define SCALE 0.125f   // 64^-0.5

typedef __attribute__((ext_vector_type(16))) __bf16 v16bf;
typedef __attribute__((ext_vector_type(8)))  float  v8f;
typedef __attribute__((ext_vector_type(4)))  unsigned int u32x4;
typedef __attribute__((ext_vector_type(8)))  int    i32x8;
typedef __attribute__((ext_vector_type(4)))  int    i32x4;
typedef __attribute__((ext_vector_type(4)))  float  f32x4;
typedef __attribute__((ext_vector_type(4)))  unsigned short u16x4;
typedef unsigned short ushort_t;

#if __has_builtin(__builtin_amdgcn_tensor_load_to_lds)
#define HAVE_TDM 1
#if __has_include(<hip/amd_detail/amd_gfx1250_TDM.h>)
#define TDM_ARGS6 1
#endif
#else
#define HAVE_TDM 0
#endif

// round-to-nearest-even fp32 -> bf16 (bit pattern)
__device__ __forceinline__ ushort_t f2bf(float x) {
  unsigned u = __float_as_uint(x);
  u += 0x7FFFu + ((u >> 16) & 1u);
  return (ushort_t)(u >> 16);
}

// A-operand fragment (16-bit, 16x32): this lane's two 8-element K chunks
// (K = kb..kb+7 and kb+8..kb+15 with kb = (lane>>4)*16). Two 16B loads.
__device__ __forceinline__ v16bf ldA(const ushort_t* p0, const ushort_t* p1) {
  union { u32x4 q[2]; v16bf v; } f;
  f.q[0] = *(const u32x4*)p0;
  f.q[1] = *(const u32x4*)p1;
  return f.v;
}

// B-operand fragment (16-bit, 32x16): 16 contiguous K values, two 16B loads.
__device__ __forceinline__ v16bf ldB(const ushort_t* p) {
  union { u32x4 q[2]; v16bf v; } f;
  f.q[0] = *(const u32x4*)p;
  f.q[1] = *(const u32x4*)(p + 8);
  return f.v;
}

__device__ __forceinline__ v8f wmma_bf16(v16bf a, v16bf b, v8f c) {
  // (neg_a, A, neg_b, B, c_mod, C, reuse_a, reuse_b)
  return __builtin_amdgcn_wmma_f32_16x16x32_bf16(false, a, false, b,
                                                 (short)0, c, false, false);
}

#if HAVE_TDM
// Issue one TDM 2-D tile load (bf16 elements) global -> LDS.
// Descriptor per CDNA5 ISA 8.3-8.5: group0 {count=1, lds_addr, global_addr, type=2},
// group1 {data_size=1(2B), tensor_dim0/1, tile_dim0/1, dim0_stride}; groups 2/3 = 0.
__device__ __forceinline__ void tdm_load_2d(unsigned lds_off, const void* gptr,
                                            unsigned tensor_d0, unsigned tensor_d1,
                                            unsigned stride0,
                                            unsigned tile_d0, unsigned tile_d1) {
  unsigned long long ga = (unsigned long long)(uintptr_t)gptr;
  u32x4 g0;
  g0[0] = 1u;                                             // count=1, no gather
  g0[1] = lds_off;                                        // LDS byte address
  g0[2] = (unsigned)(ga & 0xFFFFFFFFull);                 // global_addr[31:0]
  g0[3] = (unsigned)((ga >> 32) & 0x01FFFFFFull) | 0x80000000u; // [56:32] | type=2
  i32x8 g1;
  g1[0] = (int)(1u << 16);                                // wg_mask=0, data_size=1 (2B)
  g1[1] = (int)((tensor_d0 & 0xFFFFu) << 16);             // tensor_dim0[15:0]
  g1[2] = (int)((tensor_d0 >> 16) | ((tensor_d1 & 0xFFFFu) << 16));
  g1[3] = (int)((tensor_d1 >> 16) | (tile_d0 << 16));     // tile_dim0
  g1[4] = (int)(tile_d1);                                 // tile_dim1, tile_dim2=0
  g1[5] = (int)stride0;                                   // dim0_stride[31:0]
  g1[6] = 0;                                              // dim0_stride[47:32], dim1_stride lo
  g1[7] = 0;
  i32x4 z4 = {0, 0, 0, 0};
#if defined(TDM_ARGS6)
  i32x8 z8 = {0, 0, 0, 0, 0, 0, 0, 0};
  __builtin_amdgcn_tensor_load_to_lds(g0, g1, z4, z4, z8, 0);
#else
  __builtin_amdgcn_tensor_load_to_lds(g0, g1, z4, z4, 0);
#endif
}
__device__ __forceinline__ unsigned lds_off_of(const void* p) {
  return (unsigned)(uintptr_t)p;   // low bits of generic shared pointer = LDS offset
}
#endif

// ---------------- prep kernels ----------------
__global__ __launch_bounds__(256) void cast_x_kernel(const float* __restrict__ x,
                                                     ushort_t* __restrict__ xb, int n4) {
  int i = blockIdx.x * blockDim.x + threadIdx.x;
  if (i >= n4) return;
  f32x4 v = ((const f32x4*)x)[i];
  u16x4 o;
  o[0] = f2bf(v[0]); o[1] = f2bf(v[1]); o[2] = f2bf(v[2]); o[3] = f2bf(v[3]);
  ((u16x4*)xb)[i] = o;
}

// Wt[j][c] = (j<1024 ? Wq[c][j] : Wkv[c][j-1024]), bf16, j in [0,3072)
__global__ __launch_bounds__(256) void build_wt_kernel(const float* __restrict__ Wq,
                                                       const float* __restrict__ Wkv,
                                                       ushort_t* __restrict__ wt) {
  int idx = blockIdx.x * blockDim.x + threadIdx.x;
  if (idx >= 3 * Cc * Cc) return;
  int j = idx >> 10;
  int c = idx & (Cc - 1);
  float v = (j < Cc) ? Wq[c * Cc + j] : Wkv[c * (2 * Cc) + (j - Cc)];
  wt[idx] = f2bf(v);
}

// ---------------- fused QKV projection GEMM ----------------
// out = x[8192,1024] @ [Wq|Wkv][1024,3072] + bias, written as bf16:
//   q,k -> [b,h,n,d]   v -> transposed [b,h,d,n]
// one wave = 16 rows x 64 cols tile; 512 mtiles x 48 ntiles = 24576 waves
__global__ __launch_bounds__(256) void qkv_gemm_kernel(const ushort_t* __restrict__ xb,
                                                       const ushort_t* __restrict__ wt,
                                                       const float* __restrict__ bq,
                                                       const float* __restrict__ bkv,
                                                       ushort_t* __restrict__ qb,
                                                       ushort_t* __restrict__ kb,
                                                       ushort_t* __restrict__ vt) {
  const int wid  = threadIdx.x >> 5;
  const int gw   = blockIdx.x * 8 + wid;
  const int mt   = gw & 511;        // consecutive waves in a block share nt -> weight reuse in L2
  const int nt   = gw >> 9;
  const int lane = threadIdx.x & 31;
  const int col  = lane & 15;
  const int hi   = lane >> 4;
  const int jbase = nt * 64;

  const ushort_t* xrow = xb + (size_t)(mt * 16 + col) * Cc;

  float acc[4][8] = {};
  for (int kc = 0; kc < Cc; kc += 32) {
    if (kc + 32 < Cc) __builtin_prefetch(xrow + kc + 32, 0, 0);  // global_prefetch_b8
    v16bf af = ldA(xrow + kc + hi * 16, xrow + kc + hi * 16 + 8);
#pragma unroll
    for (int t = 0; t < 4; ++t) {
      const int j = jbase + t * 16 + col;
      v16bf bfrag = ldB(wt + (size_t)j * Cc + kc + hi * 16);
      v8f c;
#pragma unroll
      for (int i = 0; i < 8; ++i) c[i] = acc[t][i];
      c = wmma_bf16(af, bfrag, c);
#pragma unroll
      for (int i = 0; i < 8; ++i) acc[t][i] = c[i];
    }
  }

  const int b = (mt * 16) >> 10;   // 16-row tile never crosses a batch boundary
#pragma unroll
  for (int t = 0; t < 4; ++t) {
    const int j = jbase + t * 16 + col;
    const float bias = (j < Cc) ? bq[j] : bkv[j - Cc];
#pragma unroll
    for (int r = 0; r < 8; ++r) {
      const int n = (mt * 16 + r + 8 * hi) & (Nn - 1);
      const ushort_t bv = f2bf(acc[t][r] + bias);
      if (j < Cc) {                       // Q
        const int h = j >> 6, d = j & 63;
        qb[(((size_t)(b * Hh + h) * Nn) + n) * Dd + d] = bv;
      } else if (j < 2 * Cc) {            // K
        const int jj = j - Cc, h = jj >> 6, d = jj & 63;
        kb[(((size_t)(b * Hh + h) * Nn) + n) * Dd + d] = bv;
      } else {                            // V, transposed [b,h,d,n]
        const int jj = j - 2 * Cc, h = jj >> 6, d = jj & 63;
        vt[(((size_t)(b * Hh + h) * Dd) + d) * Nn + n] = bv;
      }
    }
  }
}

// ---------------- flash attention ----------------
// One block = 8 waves sharing one (b,h); each wave owns a 16-query tile.
// K (32x64) and V^T (64x32) tiles are staged into LDS by the Tensor Data Mover
// (double buffered, s_wait_tensorcnt + workgroup barriers), so K/V is fetched
// once per block instead of once per wave (8x less L2 traffic).
// S^T = K(A) x Q^T(B) -> per-lane online softmax over keys (one xor-16 shuffle)
// O^T = V^T(A) x P^T(B).
__global__ __launch_bounds__(256) void attn_kernel(const ushort_t* __restrict__ qb,
                                                   const ushort_t* __restrict__ kb,
                                                   const ushort_t* __restrict__ vt,
                                                   float* __restrict__ out) {
#if HAVE_TDM
  __shared__ ushort_t ksm[2][32 * 64];   // [buf][key][d]   4KB each
  __shared__ ushort_t vsm[2][64 * 32];   // [buf][d][key]   4KB each
#endif
  const int wid  = threadIdx.x >> 5;
  const int bh   = blockIdx.x >> 3;                 // 128 (b,h) pairs
  const int qt   = (blockIdx.x & 7) * 8 + wid;      // 64 query tiles per (b,h)
  const int lane = threadIdx.x & 31;
  const int col  = lane & 15;
  const int hi   = lane >> 4;

  const ushort_t* qp = qb + (size_t)bh * Nn * Dd;
  const ushort_t* kp = kb + (size_t)bh * Nn * Dd;
  const ushort_t* vp = vt + (size_t)bh * Dd * Nn;

  const int qrow = qt * 16 + col;
  const v16bf qf0 = ldB(qp + (size_t)qrow * Dd + hi * 16);
  const v16bf qf1 = ldB(qp + (size_t)qrow * Dd + 32 + hi * 16);

  float m_run = -3.0e38f, l_run = 0.0f;
  float acc[4][8] = {};   // O^T: 4 d-tiles of [16d x 16q], lane owns col=query

#if HAVE_TDM
  if (wid == 0) {  // prologue: stage tiles for iteration 0
    // K tile: rows are contiguous (stride 64 elements)
    tdm_load_2d(lds_off_of(&ksm[0][0]), kp, 64u, (unsigned)Nn, 64u, 64u, 32u);
    // V^T tile: 64 rows of 32 keys at stride 1024 elements
    tdm_load_2d(lds_off_of(&vsm[0][0]), vp, (unsigned)Nn, 64u, 1024u, 32u, 64u);
  }
#endif

  for (int it = 0; it < Nn / 32; ++it) {
    const int m0 = it * 32;
#if HAVE_TDM
    const int buf = it & 1;
    if (wid == 0) {
      if (it + 1 < Nn / 32) {   // issue next tiles into the other buffer, then
        tdm_load_2d(lds_off_of(&ksm[buf ^ 1][0]), kp + (size_t)(m0 + 32) * Dd,
                    64u, (unsigned)Nn, 64u, 64u, 32u);
        tdm_load_2d(lds_off_of(&vsm[buf ^ 1][0]), vp + (m0 + 32),
                    (unsigned)Nn, 64u, 1024u, 32u, 64u);
        __builtin_amdgcn_s_wait_tensorcnt(2);  // wait for the CURRENT buffer's 2 loads
      } else {
        __builtin_amdgcn_s_wait_tensorcnt(0);
      }
    }
    __syncthreads();           // current buffer ready for all 8 waves
    const ushort_t* kx0 = &ksm[buf][(size_t)col * 64];
    const ushort_t* kx1 = &ksm[buf][(size_t)(16 + col) * 64];
#else
    const ushort_t* kx0 = kp + (size_t)(m0 + col) * Dd;
    const ushort_t* kx1 = kp + (size_t)(m0 + 16 + col) * Dd;
#endif

    // ---- S^T tiles for keys [m0,m0+32) : A = K rows, B = Q^T ----
    v8f s0 = {0.f,0.f,0.f,0.f,0.f,0.f,0.f,0.f};
    v8f s1 = {0.f,0.f,0.f,0.f,0.f,0.f,0.f,0.f};
    s0 = wmma_bf16(ldA(kx0 + hi * 16,      kx0 + hi * 16 + 8), qf0, s0);
    s0 = wmma_bf16(ldA(kx0 + 32 + hi * 16, kx0 + 32 + hi * 16 + 8), qf1, s0);
    s1 = wmma_bf16(ldA(kx1 + hi * 16,      kx1 + hi * 16 + 8), qf0, s1);
    s1 = wmma_bf16(ldA(kx1 + 32 + hi * 16, kx1 + 32 + hi * 16 + 8), qf1, s1);

    float S0[8], S1[8];
#pragma unroll
    for (int r = 0; r < 8; ++r) { S0[r] = s0[r] * SCALE; S1[r] = s1[r] * SCALE; }

    // ---- online softmax (keys on M axis; lane pair covers 16+16 keys) ----
    float bm = -3.0e38f;
#pragma unroll
    for (int r = 0; r < 8; ++r) bm = fmaxf(bm, fmaxf(S0[r], S1[r]));
    bm = fmaxf(bm, __shfl_xor(bm, 16, 32));
    const float mn = fmaxf(m_run, bm);
    const float so = __expf(m_run - mn);

    float p0[8], p1[8], sum = 0.f;
#pragma unroll
    for (int r = 0; r < 8; ++r) {
      p0[r] = __expf(S0[r] - mn);
      p1[r] = __expf(S1[r] - mn);
      sum += p0[r] + p1[r];
    }
    sum += __shfl_xor(sum, 16, 32);
    l_run = l_run * so + sum;
    m_run = mn;
#pragma unroll
    for (int t = 0; t < 4; ++t)
#pragma unroll
      for (int i = 0; i < 8; ++i) acc[t][i] *= so;

    // ---- pack P^T into B-operand layout (K = m offset 0..31) ----
    union { ushort_t u[16]; v16bf v; } pf;
#pragma unroll
    for (int r = 0; r < 8; ++r) {
      const float o0 = __shfl_xor(p0[r], 16, 32);
      const float o1 = __shfl_xor(p1[r], 16, 32);
      pf.u[r]     = f2bf(hi ? o1 : p0[r]);
      pf.u[r + 8] = f2bf(hi ? p1[r] : o0);
    }

    // ---- O^T += V^T(A) x P^T(B), 4 d-tiles ----
#pragma unroll
    for (int t = 0; t < 4; ++t) {
#if HAVE_TDM
      const ushort_t* vrow = &vsm[buf][(size_t)(t * 16 + col) * 32];
#else
      const ushort_t* vrow = vp + (size_t)(t * 16 + col) * Nn + m0;
#endif
      v16bf vf = ldA(vrow + hi * 16, vrow + hi * 16 + 8);
      v8f c;
#pragma unroll
      for (int i = 0; i < 8; ++i) c[i] = acc[t][i];
      c = wmma_bf16(vf, pf.v, c);
#pragma unroll
      for (int i = 0; i < 8; ++i) acc[t][i] = c[i];
    }
#if HAVE_TDM
    __syncthreads();           // all waves done with this buffer before it is refilled
#endif
  }

  // ---- normalize and write out[b, n, h*64 + d] (fp32) ----
  const float inv_l = 1.0f / l_run;
  const int b = bh >> 4, h = bh & 15;
  float* orow = out + ((size_t)(b * Nn + qrow)) * Cc + h * Dd;
#pragma unroll
  for (int t = 0; t < 4; ++t)
#pragma unroll
    for (int r = 0; r < 8; ++r) {
      const int d = t * 16 + r + 8 * hi;   // C-layout: M = r + 8*(lane>>4)
      orow[d] = acc[t][r] * inv_l;
    }
}

// ---------------- launch ----------------
extern "C" void kernel_launch(void* const* d_in, const int* in_sizes, int n_in,
                              void* d_out, int out_size, void* d_ws, size_t ws_size,
                              hipStream_t stream) {
  (void)in_sizes; (void)n_in; (void)out_size; (void)ws_size;
  const float* x   = (const float*)d_in[0];
  const float* Wq  = (const float*)d_in[1];
  const float* bq  = (const float*)d_in[2];
  const float* Wkv = (const float*)d_in[3];
  const float* bkv = (const float*)d_in[4];
  float* out = (float*)d_out;

  char* ws = (char*)d_ws;
  // 16MB xb | 6MB Wt | 16MB q | 16MB k | 16MB vT  = 70MB total
  ushort_t* xb = (ushort_t*)(ws);
  ushort_t* wt = (ushort_t*)(ws + (size_t)(16u << 20));
  ushort_t* qb = (ushort_t*)(ws + (size_t)(22u << 20));
  ushort_t* kb = (ushort_t*)(ws + (size_t)(38u << 20));
  ushort_t* vt = (ushort_t*)(ws + (size_t)(54u << 20));

  cast_x_kernel<<<8192, 256, 0, stream>>>(x, xb, (Bb * Nn * Cc) / 4);
  build_wt_kernel<<<12288, 256, 0, stream>>>(Wq, Wkv, wt);
  // 512 mtiles * 48 ntiles = 24576 waves / 8 per block
  qkv_gemm_kernel<<<3072, 256, 0, stream>>>(xb, wt, bq, bkv, qb, kb, vt);
  // 128 (b,h) * 64 qtiles = 8192 waves / 8 per block
  attn_kernel<<<1024, 256, 0, stream>>>(qb, kb, vt, out);
}
